// GNNEdgeClassifier_52441550684387
// MI455X (gfx1250) — matrix operand order repaired and checked by reading, not compiled, not run on hardware
//
#include <hip/hip_runtime.h>
#include <hip/hip_bf16.h>

#define NN  50000
#define NNP 50048   // padded to a multiple of 64 for 64-row GEMM tiles
#define NE  800000
#define ND  128
#define HID 256
#define NL  3

typedef __attribute__((ext_vector_type(16))) __bf16 v16bf;
typedef __attribute__((ext_vector_type(8)))  float  v8f;

union Frag16 { v16bf v; uint4 q[2]; };

__device__ __forceinline__ float geluf(float x) {
    return 0.5f * x * (1.0f + erff(x * 0.70710678118654752f));
}

// ---------------- utility kernels ----------------
__global__ void k_fill0(float* p, size_t n) {
    size_t i = (size_t)blockIdx.x * blockDim.x + threadIdx.x;
    if (i < n) p[i] = 0.0f;
}

__global__ void k_cvt_bf16(const float* in, __bf16* out, size_t n) {
    size_t i = (size_t)blockIdx.x * blockDim.x + threadIdx.x;
    if (i < n) out[i] = (__bf16)in[i];
}

__global__ void k_copyrow(const float* in, float* out) {
    out[threadIdx.x] = in[threadIdx.x];
}

// Build WMMA B-fragment layout from two stacked K-major weights:
// K' = 2*D rows; k < D -> Wr[k][n], else Wt[k-D][n].
// flat index = (((ch*16 + tile)*32 + lane)*16 + i),
// k = ch*32 + (lane>>4)*16 + i, n = tile*16 + (lane&15).
__global__ void k_wfrag(const float* Wr, const float* Wt, int D, __bf16* out) {
    int idx = blockIdx.x * blockDim.x + threadIdx.x;
    int total = 2 * D * HID;
    if (idx >= total) return;
    int i    = idx & 15;
    int lane = (idx >> 4) & 31;
    int tile = (idx >> 9) & 15;
    int ch   = idx >> 13;
    int k = ch * 32 + (lane >> 4) * 16 + i;
    int n = tile * 16 + (lane & 15);
    float v = (k < D) ? Wr[k * HID + n] : Wt[(k - D) * HID + n];
    out[idx] = (__bf16)v;
}

// ---------------- scatter (segment_sum of weighted messages) ----------------
// one thread per 4 features of one edge; dshift = log2(D)
__global__ void k_scatter(const float* __restrict__ hin, const int* __restrict__ src,
                          const int* __restrict__ dst, const float* __restrict__ ew,
                          float* __restrict__ agg, int dshift) {
    size_t idx = (size_t)blockIdx.x * blockDim.x + threadIdx.x;
    int e  = (int)(idx >> (dshift - 2));
    int t4 = (int)(idx & (((size_t)1 << (dshift - 2)) - 1)) << 2;
    int s = src[e], d = dst[e];
    float w = ew[e];
    float4 v = *(const float4*)(hin + ((size_t)s << dshift) + t4);
    float* base = agg + ((size_t)d << dshift) + t4;
    atomicAdd(base + 0, v.x * w);
    atomicAdd(base + 1, v.y * w);
    atomicAdd(base + 2, v.z * w);
    atomicAdd(base + 3, v.w * w);
}

// ---------------- GraphConv GEMM: hpre = [agg | h] @ Wfrag + bias ----------------
// 64 rows x 256 cols per block; 16 waves, wave w -> col tile w, 4 row tiles each.
__global__ __launch_bounds__(512) void k_gemm_conv(const __bf16* __restrict__ Aa,
                                                   const __bf16* __restrict__ Ax, int D,
                                                   const __bf16* __restrict__ wfrag,
                                                   const float* __restrict__ bias,
                                                   float* __restrict__ out) {
    __shared__ __bf16 As[64 * 512];  // 64KB max (layer0 uses 32KB of it)
    int tid  = threadIdx.x;
    int wave = tid >> 5;
    int lane = tid & 31;
    int hlf  = lane >> 4;
    int l    = lane & 15;
    int h8   = hlf * 8;
    int rowBase = blockIdx.x * 64;
    int K2 = 2 * D;
    int qshift = (D == ND) ? 5 : 6;        // log2(uint4 per row)
    int totalQuads = 64 << qshift;

    // cooperative stage of full 64 x K2 A tile (b128 loads)
    for (int flat = tid; flat < totalQuads; flat += 512) {
        int r   = flat >> qshift;
        int qi  = flat & ((1 << qshift) - 1);
        int c16 = qi << 3;
        const __bf16* sp = (c16 < D) ? (Aa + (size_t)(rowBase + r) * D + c16)
                                     : (Ax + (size_t)(rowBase + r) * D + (c16 - D));
        ((uint4*)&As[(size_t)r * K2])[qi] = *(const uint4*)sp;
    }
    __syncthreads();

    v8f zero = {0.f,0.f,0.f,0.f,0.f,0.f,0.f,0.f};
    v8f acc[4] = {zero, zero, zero, zero};
    int nCh = 1 << (qshift - 2);           // K2/32
    for (int ch = 0; ch < nCh; ++ch) {
        Frag16 b;
        const __bf16* bp = wfrag + (((size_t)ch * 16 + wave) * 32 + lane) * 16;
        b.q[0] = ((const uint4*)bp)[0];
        b.q[1] = ((const uint4*)bp)[1];
        if (ch + 1 < nCh)
            __builtin_prefetch(wfrag + (((size_t)(ch + 1) * 16 + wave) * 32 + lane) * 16, 0, 2);
#pragma unroll
        for (int rt = 0; rt < 4; ++rt) {
            Frag16 a;
            const __bf16* ap = &As[(size_t)(rt * 16 + l) * K2 + ch * 32 + hlf * 16];
            a.q[0] = *(const uint4*)ap;
            a.q[1] = *(const uint4*)(ap + 8);
            acc[rt] = __builtin_amdgcn_wmma_f32_16x16x32_bf16(false, a.v, false, b.v,
                                                              (short)0, acc[rt], false, false);
        }
    }

    int col = wave * 16 + l;
    float bv = bias[col];
#pragma unroll
    for (int rt = 0; rt < 4; ++rt) {
#pragma unroll
        for (int r = 0; r < 8; ++r) {
            int row = rowBase + rt * 16 + r + hlf * 8;
            if (row < NN) out[(size_t)row * HID + col] = acc[rt][r] + bv;
        }
    }
}

// ---------------- GELU column stats ----------------
#define STAT_ROWS 100
__global__ void k_stats(const float* __restrict__ hpre, float* __restrict__ s1,
                        float* __restrict__ s2) {
    int c  = threadIdx.x;
    int r0 = blockIdx.x * STAT_ROWS;
    float a = 0.f, b = 0.f;
    for (int r = r0; r < r0 + STAT_ROWS; ++r) {
        float g = geluf(hpre[(size_t)r * HID + c]);
        a += g; b += g * g;
    }
    atomicAdd(&s1[c], a);
    atomicAdd(&s2[c], b);
}

// ---------------- normalize: h = BN(gelu(hpre)); also emit bf16 ----------------
__global__ void k_norm(const float* __restrict__ hpre, const float* __restrict__ s1,
                       const float* __restrict__ s2, const float* __restrict__ gamma,
                       const float* __restrict__ beta, float* __restrict__ h,
                       __bf16* __restrict__ hb) {
    int c  = threadIdx.x;
    int r0 = blockIdx.x * STAT_ROWS;
    float invN = 1.0f / (float)NN;
    float mean = s1[c] * invN;
    float var  = s2[c] * invN - mean * mean;
    float sc   = rsqrtf(var + 1e-5f) * gamma[c];
    float sh   = beta[c] - mean * sc;
    for (int r = r0; r < r0 + STAT_ROWS; ++r) {
        float g = geluf(hpre[(size_t)r * HID + c]);
        float o = g * sc + sh;
        h[(size_t)r * HID + c]  = o;
        hb[(size_t)r * HID + c] = (__bf16)o;
    }
}

// ---------------- edge classifier: 64 edges per block, 16 waves ----------------
__global__ __launch_bounds__(512) void k_edge_mlp(const __bf16* __restrict__ hb,
                                                  const int* __restrict__ src,
                                                  const int* __restrict__ dst,
                                                  const float* __restrict__ ea,
                                                  const __bf16* __restrict__ w1frag,
                                                  const float* __restrict__ w1last,
                                                  const float* __restrict__ b1,
                                                  const float* __restrict__ W2,
                                                  const float* __restrict__ b2,
                                                  float* __restrict__ out) {
    __shared__ __bf16 As[64 * 512];   // 64KB: ef rows [h[src] | h[dst]]
    __shared__ float  eaS[64];
    __shared__ float  red[64][16];
    int tid  = threadIdx.x;
    int wave = tid >> 5;
    int lane = tid & 31;
    int hlf  = lane >> 4;
    int l    = lane & 15;
    int h8   = hlf * 8;
    int e0   = blockIdx.x * 64;

    if (tid < 64) eaS[tid] = ea[e0 + tid];
    // stage 64 x 512 bf16 (4096 uint4, 8 per thread)
    for (int flat = tid; flat < 4096; flat += 512) {
        int r   = flat >> 6;
        int qi  = flat & 63;
        int c16 = qi << 3;
        int e   = e0 + r;
        int node = (c16 < HID) ? src[e] : dst[e];
        int cc   = (c16 < HID) ? c16 : c16 - HID;
        ((uint4*)&As[(size_t)r * 512])[qi] = *(const uint4*)(hb + (size_t)node * HID + cc);
    }
    __syncthreads();

    v8f zero = {0.f,0.f,0.f,0.f,0.f,0.f,0.f,0.f};
    v8f acc[4] = {zero, zero, zero, zero};
    for (int ch = 0; ch < 16; ++ch) {
        Frag16 b;
        const __bf16* bp = w1frag + (((size_t)ch * 16 + wave) * 32 + lane) * 16;
        b.q[0] = ((const uint4*)bp)[0];
        b.q[1] = ((const uint4*)bp)[1];
        if (ch < 15)
            __builtin_prefetch(w1frag + (((size_t)(ch + 1) * 16 + wave) * 32 + lane) * 16, 0, 2);
#pragma unroll
        for (int rt = 0; rt < 4; ++rt) {
            Frag16 a;
            const __bf16* ap = &As[(size_t)(rt * 16 + l) * 512 + ch * 32 + hlf * 16];
            a.q[0] = *(const uint4*)ap;
            a.q[1] = *(const uint4*)(ap + 8);
            acc[rt] = __builtin_amdgcn_wmma_f32_16x16x32_bf16(false, a.v, false, b.v,
                                                              (short)0, acc[rt], false, false);
        }
    }

    int col = wave * 16 + l;
    float bv  = b1[col];
    float wl  = w1last[col];   // W1 row 512 (edge_attr column), rank-1 update
    float w2v = W2[col];
#pragma unroll
    for (int rt = 0; rt < 4; ++rt) {
#pragma unroll
        for (int r = 0; r < 8; ++r) {
            int row = rt * 16 + r + hlf * 8;
            float hv = geluf(acc[rt][r] + bv + eaS[row] * wl);
            float v  = hv * w2v;
            // sum over the 16 lanes of this half (16 cols of this tile)
            v += __shfl_xor(v, 1);
            v += __shfl_xor(v, 2);
            v += __shfl_xor(v, 4);
            v += __shfl_xor(v, 8);
            if (l == 0) red[row][wave] = v;
        }
    }
    __syncthreads();
    if (tid < 64) {
        float s = 0.f;
#pragma unroll
        for (int w = 0; w < 16; ++w) s += red[tid][w];
        s += b2[0];
        out[e0 + tid] = 1.0f / (1.0f + expf(-s));
    }
}

// ---------------- host launcher ----------------
extern "C" void kernel_launch(void* const* d_in, const int* in_sizes, int n_in,
                              void* d_out, int out_size, void* d_ws, size_t ws_size,
                              hipStream_t stream) {
    (void)in_sizes; (void)n_in; (void)out_size; (void)ws_size;
    const float* x    = (const float*)d_in[0];
    const int*   ei   = (const int*)d_in[1];
    const float* ea   = (const float*)d_in[2];
    const float* Wr0  = (const float*)d_in[3];
    const float* br0  = (const float*)d_in[4];
    const float* Wt0  = (const float*)d_in[5];
    const float* Wr   = (const float*)d_in[6];
    const float* br   = (const float*)d_in[7];
    const float* Wt   = (const float*)d_in[8];
    const float* gam  = (const float*)d_in[9];
    const float* bet  = (const float*)d_in[10];
    const float* W1   = (const float*)d_in[11];
    const float* b1   = (const float*)d_in[12];
    const float* W2   = (const float*)d_in[13];
    const float* b2   = (const float*)d_in[14];
    float* out = (float*)d_out;
    const int* srcI = ei;
    const int* dstI = ei + NE;

    char* ws = (char*)d_ws;
    size_t off = 0;
    auto take = [&](size_t bytes) -> char* {
        char* p = ws + off;
        off += (bytes + 255) & ~(size_t)255;
        return p;
    };
    float*  agg    = (float*) take((size_t)NN * HID * 4);
    float*  hpre   = (float*) take((size_t)NN * HID * 4);
    float*  h      = (float*) take((size_t)NN * HID * 4);
    __bf16* aggb   = (__bf16*)take((size_t)NNP * HID * 2);   // padded rows
    __bf16* hb     = (__bf16*)take((size_t)NNP * HID * 2);   // padded rows
    __bf16* xb     = (__bf16*)take((size_t)NNP * ND * 2);    // padded rows
    __bf16* wfrag0 = (__bf16*)take((size_t)2 * ND * HID * 2);
    __bf16* wfragL[NL];
    for (int i = 0; i < NL; ++i) wfragL[i] = (__bf16*)take((size_t)2 * HID * HID * 2);
    __bf16* w1frag = (__bf16*)take((size_t)512 * HID * 2);
    float*  w1last = (float*) take(HID * 4);
    float*  s1     = (float*) take(HID * 4);
    float*  s2     = (float*) take(HID * 4);

    // --- weight prep (idempotent every call) ---
    k_wfrag<<<(2 * ND * HID) / 256, 256, 0, stream>>>(Wr0, Wt0, ND, wfrag0);
    for (int i = 0; i < NL; ++i)
        k_wfrag<<<(2 * HID * HID) / 256, 256, 0, stream>>>(Wr + (size_t)i * HID * HID,
                                                           Wt + (size_t)i * HID * HID,
                                                           HID, wfragL[i]);
    // W1: rows 0..511 as stacked 256+256, row 512 kept separate (edge_attr column)
    k_wfrag<<<(2 * HID * HID) / 256, 256, 0, stream>>>(W1, W1 + (size_t)HID * HID, HID, w1frag);
    k_copyrow<<<1, HID, 0, stream>>>(W1 + (size_t)512 * HID, w1last);
    k_cvt_bf16<<<((size_t)NN * ND + 255) / 256, 256, 0, stream>>>(x, xb, (size_t)NN * ND);

    // --- layer 0 (128 -> 256) ---
    k_fill0<<<((size_t)NN * ND + 255) / 256, 256, 0, stream>>>(agg, (size_t)NN * ND);
    k_scatter<<<(NE * (ND / 4)) / 256, 256, 0, stream>>>(x, srcI, dstI, ea, agg, 7);
    k_cvt_bf16<<<((size_t)NN * ND + 255) / 256, 256, 0, stream>>>(agg, aggb, (size_t)NN * ND);
    k_gemm_conv<<<NNP / 64, 512, 0, stream>>>(aggb, xb, ND, wfrag0, br0, hpre);
    k_fill0<<<2, 256, 0, stream>>>(s1, 512);   // s1 and s2 are contiguous
    k_stats<<<NN / STAT_ROWS, HID, 0, stream>>>(hpre, s1, s2);
    k_norm<<<NN / STAT_ROWS, HID, 0, stream>>>(hpre, s1, s2, gam, bet, h, hb);

    // --- layers 1..3 (256 -> 256) ---
    for (int i = 0; i < NL; ++i) {
        k_fill0<<<((size_t)NN * HID + 255) / 256, 256, 0, stream>>>(agg, (size_t)NN * HID);
        k_scatter<<<(NE * (HID / 4)) / 256, 256, 0, stream>>>(h, srcI, dstI, ea, agg, 8);
        k_cvt_bf16<<<((size_t)NN * HID + 255) / 256, 256, 0, stream>>>(agg, aggb, (size_t)NN * HID);
        k_gemm_conv<<<NNP / 64, 512, 0, stream>>>(aggb, hb, HID, wfragL[i],
                                                  br + (size_t)i * HID, hpre);
        k_fill0<<<2, 256, 0, stream>>>(s1, 512);
        k_stats<<<NN / STAT_ROWS, HID, 0, stream>>>(hpre, s1, s2);
        k_norm<<<NN / STAT_ROWS, HID, 0, stream>>>(hpre, s1, s2,
                                                   gam + (size_t)(i + 1) * HID,
                                                   bet + (size_t)(i + 1) * HID, h, hb);
    }

    // --- edge classifier ---
    k_edge_mlp<<<NE / 64, 512, 0, stream>>>(hb, srcI, dstI, ea, w1frag, w1last,
                                            b1, W2, b2, out);
}
